// NnBoard768_36498632081976
// MI455X (gfx1250) — compile-verified, matrix operand order checked by reading, not run on hardware
//
#include <hip/hip_runtime.h>
#include <math.h>

#define N_FEAT   768
#define FT_OUT   512
#define NNZ_PER  32
#define TM       32      // positions per workgroup (2 WMMA M-tiles)
#define ROWS     776     // padded board row stride in bf16 elems (bank-conflict-free, 16B-mult)
#define NCHUNKS  32      // FT_OUT / 16
#define KSTEPS   24      // N_FEAT / 32

typedef __attribute__((ext_vector_type(16))) __bf16 v16bf;
typedef __attribute__((ext_vector_type(8)))  float  v8f;

__device__ __forceinline__ unsigned short f32_to_bf16(float f) {
    unsigned int u = __float_as_uint(f);
    unsigned int r = u + 0x7FFFu + ((u >> 16) & 1u);   // round-to-nearest-even
    return (unsigned short)(r >> 16);
}
__device__ __forceinline__ float bf16_to_f32(unsigned short h) {
    return __uint_as_float(((unsigned int)h) << 16);
}

// ---------------------------------------------------------------------------
// Pack ft_w (fp32 [FT_OUT][N_FEAT]) into bf16 B-fragments in the exact 32x16
// WMMA B-operand register image (16-bit layout, §7.12.2):
//   frag(nc,ks): lane L holds N = nc*16 + (L&15);
//     half0 (VGPR0-3): K pairs {k0+koff+2j, +1}, koff = (L<16?0:8)
//     half1 (VGPR4-7): K pairs {k0+16+koff+2j, +1}
//   storage: pack[frag][half][lane][16B]  -> main kernel loads 2 coalesced b128
// ---------------------------------------------------------------------------
__global__ void pack_weights(const float* __restrict__ ft_w,
                             unsigned int* __restrict__ pack) {
    int d = blockIdx.x * blockDim.x + threadIdx.x;      // dword index
    if (d >= NCHUNKS * KSTEPS * 256) return;            // 256 dwords per frag
    int frag = d >> 8;
    int rem  = d & 255;
    int half = rem >> 7;
    int lane = (rem >> 2) & 31;
    int j    = rem & 3;
    int nc = frag / KSTEPS;
    int ks = frag % KSTEPS;
    int n  = nc * 16 + (lane & 15);
    int k  = ks * 32 + half * 16 + ((lane < 16) ? 0 : 8) + 2 * j;
    unsigned int lo = f32_to_bf16(ft_w[n * N_FEAT + k]);
    unsigned int hi = f32_to_bf16(ft_w[n * N_FEAT + k + 1]);
    pack[d] = lo | (hi << 16);
}

// ---------------------------------------------------------------------------
// Main kernel: 32 positions per workgroup, 8 waves.
// LDS boards (bf16, padded rows) -> scatter -> dense bf16 WMMA GEMM -> clip
// -> out_w dot (shfl reduce + ds_add_f32) -> sigmoid.
// ---------------------------------------------------------------------------
__global__ __launch_bounds__(256) void nnue_kernel(
    const int*   __restrict__ stm_idx,
    const int*   __restrict__ nstm_idx,
    const float* __restrict__ values,
    const unsigned int* __restrict__ pack,
    const float* __restrict__ ft_b,
    const float* __restrict__ out_w,
    const float* __restrict__ out_b,
    float*       __restrict__ out)
{
    __shared__ unsigned short s_board[2 * TM * ROWS];   // 97 KB, bank-padded
    __shared__ float          s_part[TM];

    const int tid    = threadIdx.x;
    const int lane   = tid & 31;
    const int wave   = tid >> 5;
    const int ln     = lane & 15;
    const int half16 = lane >> 4;
    const int P      = blockIdx.x * TM;

    // ---- zero boards + partials ----
    unsigned int* bz = (unsigned int*)s_board;          // TM*ROWS dwords total
    for (int i = tid; i < TM * ROWS; i += 256) bz[i] = 0u;
    if (tid < TM) s_part[tid] = 0.0f;
    __syncthreads();

    // ---- race-free scatter: one owner thread per (perspective,row) ----
    if (tid < 64) {
        int persp = tid >> 5;
        int pos   = tid & 31;
        const int* idx = persp ? nstm_idx : stm_idx;
        unsigned short* row = &s_board[persp * TM * ROWS + pos * ROWS];
        int base = (P + pos) * NNZ_PER;
        for (int i = 0; i < NNZ_PER; ++i) {
            int   f = idx[(base + i) * 2 + 1];          // feature id
            float v = values[base + i];
            row[f] = f32_to_bf16(bf16_to_f32(row[f]) + v);  // duplicates sum
        }
    }
    __syncthreads();

    // ---- GEMM: each wave covers 4 output chunks; per chunk 4 C-tiles ----
    union alignas(32) AB { unsigned int u[8]; v16bf v; };

    for (int it = 0; it < 4; ++it) {
        int nc = wave + it * 8;                         // output chunk (16 cols)
        float bias = ft_b[nc * 16 + ln];
        v8f acc[4];                                     // [persp*2 + mtile]
        #pragma unroll
        for (int t = 0; t < 4; ++t)
            #pragma unroll
            for (int r = 0; r < 8; ++r) acc[t][r] = bias;

        for (int ks = 0; ks < KSTEPS; ++ks) {
            int k0 = ks * 32;
            // B fragment: two coalesced 16B loads from the packed image (L2-resident)
            AB bfr;
            const uint4* bp = (const uint4*)(pack + (nc * KSTEPS + ks) * 256);
            *(uint4*)&bfr.u[0] = bp[lane];
            *(uint4*)&bfr.u[4] = bp[32 + lane];
            #pragma unroll
            for (int t = 0; t < 4; ++t) {               // 4 wmma per B load
                int persp = t >> 1, mt = t & 1;
                const unsigned short* ap =
                    &s_board[persp * TM * ROWS + (mt * 16 + ln) * ROWS
                             + k0 + half16 * 8];
                AB afr;                                 // A frag: 2 x ds_load_b128
                *(uint4*)&afr.u[0] = *(const uint4*)ap;
                *(uint4*)&afr.u[4] = *(const uint4*)(ap + 16);
                acc[t] = __builtin_amdgcn_wmma_f32_16x16x32_bf16(
                    false, afr.v, false, bfr.v, (short)0, acc[t], false, false);
            }
        }

        // ---- epilogue: clip, scale by out_w, reduce across the 16 n-lanes ----
        #pragma unroll
        for (int t = 0; t < 4; ++t) {
            int persp = t >> 1, mt = t & 1;
            float w = out_w[persp * FT_OUT + nc * 16 + ln];
            float s[8];
            #pragma unroll
            for (int r = 0; r < 8; ++r) {
                float h = acc[t][r];
                h = h < 0.0f ? 0.0f : (h > 1.0f ? 1.0f : h);
                s[r] = h * w;
            }
            #pragma unroll
            for (int m = 1; m < 16; m <<= 1)
                #pragma unroll
                for (int r = 0; r < 8; ++r)
                    s[r] += __shfl_xor(s[r], m, 32);
            if (ln == 0) {                              // lanes 0 and 16 write
                int mbase = mt * 16 + half16 * 8;
                #pragma unroll
                for (int r = 0; r < 8; ++r)
                    atomicAdd(&s_part[mbase + r], s[r]);   // ds_add_f32
            }
        }
    }
    __syncthreads();

    if (tid < TM) {
        float z = s_part[tid] + out_b[0];
        out[P + tid] = 1.0f / (1.0f + expf(-z));
    }
}

extern "C" void kernel_launch(void* const* d_in, const int* in_sizes, int n_in,
                              void* d_out, int out_size, void* d_ws, size_t ws_size,
                              hipStream_t stream) {
    const int*   stm    = (const int*)  d_in[0];
    const int*   nstm   = (const int*)  d_in[1];
    const float* values = (const float*)d_in[2];
    // d_in[3] = size scalar (compile-time constant here)
    const float* ft_w   = (const float*)d_in[4];
    const float* ft_b   = (const float*)d_in[5];
    const float* out_w  = (const float*)d_in[6];
    const float* out_b  = (const float*)d_in[7];

    unsigned int* pack = (unsigned int*)d_ws;           // 768 KB packed bf16 weights

    const int packDwords = NCHUNKS * KSTEPS * 256;      // 196608
    pack_weights<<<(packDwords + 255) / 256, 256, 0, stream>>>(ft_w, pack);

    const int B = out_size;                             // 65536
    nnue_kernel<<<B / TM, 256, 0, stream>>>(stm, nstm, values, pack,
                                            ft_b, out_w, out_b, (float*)d_out);
}
